// TransConv_54829552500960
// MI455X (gfx1250) — matrix operand order repaired and checked by reading, not compiled, not run on hardware
//
#include <hip/hip_runtime.h>
#include <hip/hip_bf16.h>
#include <stdint.h>

// ---------------------------------------------------------------------------
// TransConv (SGFormer-style linear attention) for MI455X / gfx1250.
// - All GEMMs: v_wmma_f32_16x16x32_bf16, wave32, register-blocked 4 column
//   tiles per wave (1 A-frag load feeds 4 WMMAs -> 4x less A traffic).
// - Row-major epilogues use a per-wave 16x68 LDS tile transpose so each lane
//   stores a 128B contiguous row segment (C-fragment is column-per-lane).
// - K^T V reduction double-buffers its B stream through LDS with
//   global_load_async_to_lds_b128 + s_wait_asynccnt (ASYNCcnt is in-order).
// ---------------------------------------------------------------------------

typedef __bf16 bf16_t;
typedef __attribute__((ext_vector_type(16))) __bf16 v16bf;
typedef __attribute__((ext_vector_type(8)))  __bf16 v8bf;
typedef __attribute__((ext_vector_type(8)))  float  v8f;
typedef __attribute__((ext_vector_type(4)))  float  v4f;

#define NROWS 65536
#define HIDD  512
#define HT    1024          // HEADS * HID
#define NF    65536.0f
#define LN_EPS 1e-5f

// ---------------- device helpers ----------------

__device__ __forceinline__ float wave_sum(float v) {
#pragma unroll
  for (int off = 16; off > 0; off >>= 1) v += __shfl_xor(v, off, 32);
  return v;
}

// A fragment, 16x32 bf16: lanes 0-15 K=k0+0..15,
// lanes 16-31 K=k0+16..31; row = lane%16.
__device__ __forceinline__ v16bf load_a(const bf16_t* __restrict__ A, size_t lda,
                                        int row, int k0, int half) {
  const bf16_t* p = A + (size_t)row * lda + (size_t)(k0 + 16 * half);
  v8bf a0 = *(const v8bf*)p;
  v8bf a1 = *(const v8bf*)(p + 8);
  return __builtin_shufflevector(a0, a1, 0,1,2,3,4,5,6,7,8,9,10,11,12,13,14,15);
}

// B fragment, 32x16 bf16 from a transposed matrix Bt[col][k]:
// lanes 0-15 K=k0..k0+15, lanes 16-31 K=k0+16..k0+31, col = lane%16.
__device__ __forceinline__ v16bf load_b(const bf16_t* __restrict__ Bt, size_t ldb,
                                        int col, int k0, int half) {
  return *(const v16bf*)(Bt + (size_t)col * ldb + (size_t)(k0 + 16 * half));
}

__device__ __forceinline__ v8f wmma_bf16(v16bf a, v16bf b, v8f c) {
  return __builtin_amdgcn_wmma_f32_16x16x32_bf16(false, a, false, b, (short)0, c,
                                                 false, false);
}

// ---------------- small utility kernels ----------------

// in: f32 [K][Nc] row-major -> out: bf16 [Nc][K] (transposed)
__global__ void transpose_cvt_kernel(const float* __restrict__ in,
                                     bf16_t* __restrict__ out, int K, int Nc) {
  size_t idx = (size_t)blockIdx.x * blockDim.x + threadIdx.x;
  size_t total = (size_t)K * Nc;
  if (idx >= total) return;
  int n = (int)(idx / K);
  int k = (int)(idx % K);
  out[idx] = (bf16_t)in[(size_t)k * Nc + n];
}

__global__ void cvt_f32_bf16_kernel(const float* __restrict__ in,
                                    bf16_t* __restrict__ out, size_t n) {
  size_t i = (size_t)blockIdx.x * blockDim.x + threadIdx.x;
  size_t stride = (size_t)gridDim.x * blockDim.x;
  for (; i < n; i += stride) out[i] = (bf16_t)in[i];
}

// inv = 1 / (||Q||_F * ||K||_F) = rsqrt(sum_q2 * sum_k2)
__global__ void finalize_inv_kernel(float* __restrict__ scal) {
  if (threadIdx.x == 0) scal[2] = rsqrtf(scal[0] * scal[1]);
}

// ---------------- GEMM: fc0 (C = A @ Bt^T + bias, f32 out), CB=4 ------------

__global__ void gemm_bias_kernel(const bf16_t* __restrict__ A,
                                 const bf16_t* __restrict__ Bt,
                                 const float* __restrict__ bias,
                                 float* __restrict__ C, int M, int Nc, int K) {
  __shared__ float lt[8][16][68];
  int ws   = threadIdx.x >> 5;
  int wid  = blockIdx.x * (blockDim.x >> 5) + ws;
  int lane = threadIdx.x & 31;
  int groups_n = Nc >> 6;                 // 64 columns per wave
  int tm = wid / groups_n, tg = wid % groups_n;
  if (tm >= (M >> 4)) return;             // wave-uniform guard
  int row0 = tm << 4, col0 = tg << 6;
  int half = lane >> 4, r = lane & 15;

  v8f c[4] = {{}, {}, {}, {}};
#pragma unroll 2
  for (int k0 = 0; k0 < K; k0 += 32) {
    v16bf a = load_a(A, (size_t)K, row0 + r, k0, half);
#pragma unroll
    for (int i = 0; i < 4; ++i) {
      v16bf b = load_b(Bt, (size_t)K, col0 + 16 * i + r, k0, half);
      c[i] = wmma_bf16(a, b, c[i]);
    }
  }
#pragma unroll
  for (int i = 0; i < 4; ++i) {
    float bv = bias[col0 + 16 * i + r];
#pragma unroll
    for (int j = 0; j < 8; ++j) lt[ws][j + 8 * half][16 * i + r] = c[i][j] + bv;
  }
  // lane (r, half) stores row r, cols [32*half, 32*half+32): 128B contiguous
  float* dst = C + (size_t)(row0 + r) * Nc + col0 + 32 * half;
#pragma unroll
  for (int q = 0; q < 8; ++q) {
    v4f t;
#pragma unroll
    for (int e = 0; e < 4; ++e) t[e] = lt[ws][r][32 * half + 4 * q + e];
    *(v4f*)(dst + 4 * q) = t;
  }
}

// ---------------- GEMM: Q/K/V projections with fused epilogues, CB=4 ---------
// mode 0 (Q): row-major bf16 out (LDS transpose) + Frobenius sumsq atomic
// mode 1 (K): transposed bf16 out [HT][N] (v8bf stores) + sumsq + ksum
// mode 2 (V): transposed bf16 out [HT][N] + row-major f32 out (LDS transpose)

__global__ void gemm_qkv_kernel(const bf16_t* __restrict__ A,
                                const bf16_t* __restrict__ Wt,
                                const float* __restrict__ bias,
                                bf16_t* __restrict__ out_row,   // mode 0
                                bf16_t* __restrict__ out_tr,    // modes 1,2
                                float*  __restrict__ out_f32,   // mode 2
                                float*  __restrict__ sq_accum,  // modes 0,1
                                float*  __restrict__ ksum,      // mode 1
                                int M, int Nc, int K, int mode) {
  __shared__ float lt[8][16][68];
  int ws   = threadIdx.x >> 5;
  int wid  = blockIdx.x * (blockDim.x >> 5) + ws;
  int lane = threadIdx.x & 31;
  int groups_n = Nc >> 6;
  int tm = wid / groups_n, tg = wid % groups_n;
  if (tm >= (M >> 4)) return;
  int row0 = tm << 4, col0 = tg << 6;
  int half = lane >> 4, r = lane & 15;

  v8f c[4] = {{}, {}, {}, {}};
#pragma unroll 2
  for (int k0 = 0; k0 < K; k0 += 32) {
    v16bf a = load_a(A, (size_t)K, row0 + r, k0, half);
#pragma unroll
    for (int i = 0; i < 4; ++i) {
      v16bf b = load_b(Wt, (size_t)K, col0 + 16 * i + r, k0, half);
      c[i] = wmma_bf16(a, b, c[i]);
    }
  }

  float v[4][8];
#pragma unroll
  for (int i = 0; i < 4; ++i) {
    float bv = bias[col0 + 16 * i + r];
#pragma unroll
    for (int j = 0; j < 8; ++j) v[i][j] = c[i][j] + bv;
  }

  if (mode == 0) {
    // LDS transpose -> 64B contiguous row-major bf16 stores per lane
#pragma unroll
    for (int i = 0; i < 4; ++i)
#pragma unroll
      for (int j = 0; j < 8; ++j) lt[ws][j + 8 * half][16 * i + r] = v[i][j];
    bf16_t* dst = out_row + (size_t)(row0 + r) * Nc + col0 + 32 * half;
#pragma unroll
    for (int q = 0; q < 4; ++q) {
      v8bf pack;
#pragma unroll
      for (int e = 0; e < 8; ++e) pack[e] = (bf16_t)lt[ws][r][32 * half + 8 * q + e];
      *(v8bf*)(dst + 8 * q) = pack;
    }
  } else {
    // transposed layout: lane owns 8 consecutive rows of one column -> 16B store
#pragma unroll
    for (int i = 0; i < 4; ++i) {
      v8bf pack;
#pragma unroll
      for (int j = 0; j < 8; ++j) pack[j] = (bf16_t)v[i][j];
      *(v8bf*)(out_tr + (size_t)(col0 + 16 * i + r) * M + row0 + 8 * half) = pack;
    }
  }
  if (mode == 2) {             // V also kept f32 row-major for the n*V term
#pragma unroll
    for (int i = 0; i < 4; ++i)
#pragma unroll
      for (int j = 0; j < 8; ++j) lt[ws][j + 8 * half][16 * i + r] = v[i][j];
    float* dst = out_f32 + (size_t)(row0 + r) * Nc + col0 + 32 * half;
#pragma unroll
    for (int q = 0; q < 8; ++q) {
      v4f t;
#pragma unroll
      for (int e = 0; e < 4; ++e) t[e] = lt[ws][r][32 * half + 4 * q + e];
      *(v4f*)(dst + 4 * q) = t;
    }
  }
  if (mode == 0 || mode == 1) {  // Frobenius norm accumulation
    float s2 = 0.f;
#pragma unroll
    for (int i = 0; i < 4; ++i)
#pragma unroll
      for (int j = 0; j < 8; ++j) s2 += v[i][j] * v[i][j];
    s2 = wave_sum(s2);
    if (lane == 0) atomicAdd(sq_accum, s2);
  }
  if (mode == 1) {               // ksum[m] = sum_n K[n,m]
#pragma unroll
    for (int i = 0; i < 4; ++i) {
      float cs = 0.f;
#pragma unroll
      for (int j = 0; j < 8; ++j) cs += v[i][j];
      cs += __shfl_down(cs, 16, 32);
      if (half == 0) atomicAdd(&ksum[col0 + 16 * i + r], cs);
    }
  }
}

// ---------------- KVT[h][d][m] = sum_l V[l,h,d]*K[l,h,m] ---------------------
// Split-K with f32 atomics; 4 d-tiles per wave share one async-LDS B stream.

__global__ void kv_gemm_kernel(const bf16_t* __restrict__ Vt,  // [HT][N]
                               const bf16_t* __restrict__ Kt,  // [HT][N]
                               float* __restrict__ KVT,        // [2][512][512]
                               int Nred, int kchunks) {
  __shared__ __align__(32) bf16_t kbuf[8][2][512];   // 8 waves x 2 bufs x 1KB
  int ws   = threadIdx.x >> 5;
  int wid  = blockIdx.x * (blockDim.x >> 5) + ws;
  int lane = threadIdx.x & 31;
  int total = 2 * 8 * 32 * kchunks;      // heads x d-groups(64) x m-tiles x chunks
  if (wid >= total) return;
  int chunk = wid % kchunks;
  int t     = wid / kchunks;
  int h     = t / (8 * 32);
  int tt    = t % (8 * 32);
  int tdg   = tt / 32;        // d group of 64
  int tmc   = tt % 32;        // m tile
  int half = lane >> 4, r = lane & 15;

  int klen = Nred / kchunks;
  int kbeg = chunk * klen;
  int kend = kbeg + klen;
  int brow = h * HIDD + tmc * 16 + r;   // Kt row (B operand, m index)

  const bf16_t* grow = Kt + (size_t)brow * Nred;
  uint32_t lbase = (uint32_t)(uintptr_t)(&kbuf[ws][0][0]);

  auto issue_async = [&](int buf, int k0) {
    uint32_t laddr = lbase + (uint32_t)buf * 1024u + (uint32_t)lane * 32u;
    uint64_t ga    = (uint64_t)(uintptr_t)(grow + k0 + 16 * half);
    asm volatile("global_load_async_to_lds_b128 %0, %1, off\n\t"
                 "global_load_async_to_lds_b128 %2, %3, off"
                 :: "v"(laddr), "v"(ga), "v"(laddr + 16u), "v"(ga + 16u)
                 : "memory");
  };

  v8f c[4] = {{}, {}, {}, {}};
  issue_async(0, kbeg);
  for (int k0 = kbeg; k0 < kend; k0 += 32) {
    int  cur  = ((k0 - kbeg) >> 5) & 1;
    bool more = (k0 + 32) < kend;
    if (more) {
      issue_async(cur ^ 1, k0 + 32);
      // in-order ASYNCcnt: <=2 outstanding means current buffer is complete
      asm volatile("s_wait_asynccnt 0x2" ::: "memory");
    } else {
      asm volatile("s_wait_asynccnt 0x0" ::: "memory");
    }
    v16bf b = *(const v16bf*)(&kbuf[ws][cur][lane * 16]);
#pragma unroll
    for (int i = 0; i < 4; ++i) {
      int arow = h * HIDD + tdg * 64 + 16 * i + r;   // Vt row (A operand, d)
      const bf16_t* pa = Vt + (size_t)arow * Nred + (size_t)(k0 + 16 * half);
      if (i == 0) __builtin_prefetch(pa + 256, 0, 1);
      v8bf a0 = *(const v8bf*)pa;
      v8bf a1 = *(const v8bf*)(pa + 8);
      v16bf a = __builtin_shufflevector(a0, a1,
                  0,1,2,3,4,5,6,7,8,9,10,11,12,13,14,15);
      c[i] = wmma_bf16(a, b, c[i]);
    }
  }
#pragma unroll
  for (int i = 0; i < 4; ++i)
#pragma unroll
    for (int j = 0; j < 8; ++j) {
      int d = tdg * 64 + 16 * i + j + 8 * half;
      int m = tmc * 16 + r;
      atomicAdd(&KVT[((size_t)h * HIDD + d) * HIDD + m], c[i][j]);
    }
}

// ---------------- denom[n,h] = inv * dot(Q[n,h,:], ksum[h,:]) + N ------------

__global__ void denom_kernel(const bf16_t* __restrict__ Q,
                             const float* __restrict__ ksum,
                             const float* __restrict__ scal,
                             float* __restrict__ denom, int Nrows) {
  int wid  = blockIdx.x * (blockDim.x >> 5) + (threadIdx.x >> 5);
  int lane = threadIdx.x & 31;
  if (wid >= Nrows * 2) return;
  int n = wid >> 1, h = wid & 1;
  v16bf qv = *(const v16bf*)(Q + (size_t)n * HT + h * HIDD + lane * 16);
  const float* ks = ksum + h * HIDD + lane * 16;
  float s = 0.f;
#pragma unroll
  for (int i = 0; i < 16; ++i) s += (float)qv[i] * ks[i];
  s = wave_sum(s);
  if (lane == 0) denom[wid] = scal[2] * s + NF;
}

// ---------------- attention GEMM + head mean + residual (pre-LN), CB=4 -------
// sum_h (inv*c_h + NF*V_h)/dn_h = sum_h inv*c_h/dn_h  (C-layout, transposed)
//                               + NF * sum_h V_h/dn_h (row-contiguous after LDS)

__global__ void attn_final_kernel(const bf16_t* __restrict__ Q,     // [N][HT]
                                  const bf16_t* __restrict__ KVTb,  // [2][512][512]
                                  const float* __restrict__ Vf32,   // [N][HT]
                                  const float* __restrict__ denom,  // [N][2]
                                  const float* __restrict__ scal,
                                  const float* __restrict__ prev,   // [N][512]
                                  float* __restrict__ outp, int M) {
  __shared__ float lt[8][16][68];
  int ws   = threadIdx.x >> 5;
  int wid  = blockIdx.x * (blockDim.x >> 5) + ws;
  int lane = threadIdx.x & 31;
  int tm = wid / 8, tg = wid % 8;
  if (tm >= (M >> 4)) return;
  int row0 = tm << 4, col0 = tg << 6;
  int half = lane >> 4, r = lane & 15;
  float inv = scal[2];

  float tacc[4][8];
#pragma unroll
  for (int i = 0; i < 4; ++i)
#pragma unroll
    for (int j = 0; j < 8; ++j) tacc[i][j] = 0.f;

#pragma unroll
  for (int h = 0; h < 2; ++h) {
    const bf16_t* Ah = Q + (size_t)h * HIDD;                 // lda = HT
    const bf16_t* Bh = KVTb + (size_t)h * HIDD * HIDD;       // [d][m], ldb = 512
    v8f c[4] = {{}, {}, {}, {}};
#pragma unroll 2
    for (int k0 = 0; k0 < HIDD; k0 += 32) {
      v16bf a = load_a(Ah, (size_t)HT, row0 + r, k0, half);
#pragma unroll
      for (int i = 0; i < 4; ++i) {
        v16bf b = load_b(Bh, (size_t)HIDD, col0 + 16 * i + r, k0, half);
        c[i] = wmma_bf16(a, b, c[i]);
      }
    }
#pragma unroll
    for (int j = 0; j < 8; ++j) {
      float rdn = inv / denom[(size_t)(row0 + j + 8 * half) * 2 + h];
#pragma unroll
      for (int i = 0; i < 4; ++i) tacc[i][j] += c[i][j] * rdn;
    }
  }
  // transpose WMMA part
#pragma unroll
  for (int i = 0; i < 4; ++i)
#pragma unroll
    for (int j = 0; j < 8; ++j) lt[ws][j + 8 * half][16 * i + r] = tacc[i][j];

  // row-contiguous phase: lane (r, half) owns row n, cols [32*half, +32)
  int n = row0 + r;
  float rd0 = NF / denom[(size_t)n * 2 + 0];
  float rd1 = NF / denom[(size_t)n * 2 + 1];
  const float* v0p = Vf32 + (size_t)n * HT + col0 + 32 * half;        // head 0
  const float* v1p = v0p + HIDD;                                      // head 1
  const float* pp  = prev + (size_t)n * HIDD + col0 + 32 * half;
  float*       op  = outp + (size_t)n * HIDD + col0 + 32 * half;
#pragma unroll
  for (int q = 0; q < 8; ++q) {
    v4f tv0 = *(const v4f*)(v0p + 4 * q);
    v4f tv1 = *(const v4f*)(v1p + 4 * q);
    v4f tpp = *(const v4f*)(pp + 4 * q);
    v4f to;
#pragma unroll
    for (int e = 0; e < 4; ++e) {
      float s = lt[ws][r][32 * half + 4 * q + e] + tv0[e] * rd0 + tv1[e] * rd1;
      to[e] = 0.25f * s + 0.5f * tpp[e];   // ALPHA=0.5, mean over 2 heads
    }
    *(v4f*)(op + 4 * q) = to;
  }
}

// ---------------- LayerNorm (one wave per 512-wide row) ----------------------

__global__ void layernorm_kernel(const float* __restrict__ in,
                                 const float* __restrict__ g,
                                 const float* __restrict__ b,
                                 float* __restrict__ outf,
                                 bf16_t* __restrict__ outbf,
                                 int relu, int Nrows) {
  int wid  = blockIdx.x * (blockDim.x >> 5) + (threadIdx.x >> 5);
  int lane = threadIdx.x & 31;
  if (wid >= Nrows) return;
  const float* row = in + (size_t)wid * HIDD + lane * 16;
  float v[16];
#pragma unroll
  for (int q = 0; q < 4; ++q) {
    v4f t = *(const v4f*)(row + q * 4);
#pragma unroll
    for (int e = 0; e < 4; ++e) v[q * 4 + e] = t[e];
  }
  float s = 0.f;
#pragma unroll
  for (int i = 0; i < 16; ++i) s += v[i];
  s = wave_sum(s);
  float mu = s * (1.0f / HIDD);
  float s2 = 0.f;
#pragma unroll
  for (int i = 0; i < 16; ++i) { float d = v[i] - mu; s2 += d * d; }
  s2 = wave_sum(s2);
  float rstd = rsqrtf(s2 * (1.0f / HIDD) + LN_EPS);
#pragma unroll
  for (int i = 0; i < 16; ++i) {
    int col = lane * 16 + i;
    float o = (v[i] - mu) * rstd * g[col] + b[col];
    if (relu) o = fmaxf(o, 0.f);
    v[i] = o;
  }
  float* orow = outf + (size_t)wid * HIDD + lane * 16;
#pragma unroll
  for (int q = 0; q < 4; ++q) {
    v4f t;
#pragma unroll
    for (int e = 0; e < 4; ++e) t[e] = v[q * 4 + e];
    *(v4f*)(orow + q * 4) = t;
  }
  if (outbf) {
    bf16_t* brow = outbf + (size_t)wid * HIDD + lane * 16;
    v8bf p0, p1;
#pragma unroll
    for (int i = 0; i < 8; ++i) { p0[i] = (bf16_t)v[i]; p1[i] = (bf16_t)v[8 + i]; }
    *(v8bf*)brow = p0;
    *(v8bf*)(brow + 8) = p1;
  }
}

// ---------------------------------------------------------------------------

extern "C" void kernel_launch(void* const* d_in, const int* in_sizes, int n_in,
                              void* d_out, int out_size, void* d_ws, size_t ws_size,
                              hipStream_t stream) {
  const float* x     = (const float*)d_in[0];
  const float* fc0_w = (const float*)d_in[1];
  const float* fc0_b = (const float*)d_in[2];
  const float* ln0_g = (const float*)d_in[3];
  const float* ln0_b = (const float*)d_in[4];
  const float* wq    = (const float*)d_in[5];
  const float* bq    = (const float*)d_in[6];
  const float* wk    = (const float*)d_in[7];
  const float* bk    = (const float*)d_in[8];
  const float* wv    = (const float*)d_in[9];
  const float* bv    = (const float*)d_in[10];
  const float* ln_g  = (const float*)d_in[11];
  const float* ln_b  = (const float*)d_in[12];

  char* ws = (char*)d_ws;
  size_t cur = 0;
  auto alloc = [&](size_t bytes) -> void* {
    void* p = ws + cur;
    cur += (bytes + 255) & ~(size_t)255;
    return p;
  };

  bf16_t* x_bf   = (bf16_t*)alloc((size_t)NROWS * HIDD * 2);
  bf16_t* fc0_wt = (bf16_t*)alloc((size_t)HIDD * HIDD * 2);
  bf16_t *wqt[2], *wkt[2], *wvt[2];
  for (int l = 0; l < 2; ++l) {
    wqt[l] = (bf16_t*)alloc((size_t)HT * HIDD * 2);
    wkt[l] = (bf16_t*)alloc((size_t)HT * HIDD * 2);
    wvt[l] = (bf16_t*)alloc((size_t)HT * HIDD * 2);
  }
  float*  prevA    = (float*) alloc((size_t)NROWS * HIDD * 4);
  bf16_t* prevA_bf = (bf16_t*)alloc((size_t)NROWS * HIDD * 2);
  float*  prevB    = (float*) alloc((size_t)NROWS * HIDD * 4);
  bf16_t* prevB_bf = (bf16_t*)alloc((size_t)NROWS * HIDD * 2);
  float*  tmp      = (float*) alloc((size_t)NROWS * HIDD * 4);
  bf16_t* Qbf      = (bf16_t*)alloc((size_t)NROWS * HT * 2);
  bf16_t* Ktb      = (bf16_t*)alloc((size_t)NROWS * HT * 2);
  bf16_t* Vtb      = (bf16_t*)alloc((size_t)NROWS * HT * 2);
  float*  Vf       = (float*) alloc((size_t)NROWS * HT * 4);
  size_t zbytes    = 256 + 4096 + (size_t)2 * HIDD * HIDD * 4;
  char*   zreg     = (char*)  alloc(zbytes);
  float*  scal     = (float*)zreg;                // [0]=sumQ2 [1]=sumK2 [2]=inv
  float*  ksum     = (float*)(zreg + 256);        // [HT]
  float*  kvtf     = (float*)(zreg + 256 + 4096); // [2][512][512]
  bf16_t* kvtb     = (bf16_t*)alloc((size_t)2 * HIDD * HIDD * 2);
  float*  denom    = (float*) alloc((size_t)NROWS * 2 * 4);

  dim3 blk(256);  // 8 wave32 per block

  // --- weight prep (transposed bf16) + x bf16
  transpose_cvt_kernel<<<(HIDD * HIDD + 255) / 256, blk, 0, stream>>>(fc0_w, fc0_wt, HIDD, HIDD);
  for (int l = 0; l < 2; ++l) {
    size_t woff = (size_t)l * HIDD * HT;
    transpose_cvt_kernel<<<(HIDD * HT + 255) / 256, blk, 0, stream>>>(wq + woff, wqt[l], HIDD, HT);
    transpose_cvt_kernel<<<(HIDD * HT + 255) / 256, blk, 0, stream>>>(wk + woff, wkt[l], HIDD, HT);
    transpose_cvt_kernel<<<(HIDD * HT + 255) / 256, blk, 0, stream>>>(wv + woff, wvt[l], HIDD, HT);
  }
  cvt_f32_bf16_kernel<<<4096, blk, 0, stream>>>(x, x_bf, (size_t)NROWS * HIDD);

  // --- fc0 + LN + ReLU
  gemm_bias_kernel<<<(NROWS / 16) * (HIDD / 64) / 8, blk, 0, stream>>>(
      x_bf, fc0_wt, fc0_b, tmp, NROWS, HIDD, HIDD);
  layernorm_kernel<<<NROWS / 8, blk, 0, stream>>>(tmp, ln0_g, ln0_b, prevA, prevA_bf, 1, NROWS);

  float*  prevf = prevA;
  bf16_t* prevb = prevA_bf;
  for (int l = 0; l < 2; ++l) {
    (void)hipMemsetAsync(zreg, 0, zbytes, stream);

    int qkv_blocks = (NROWS / 16) * (HT / 64) / 8;
    gemm_qkv_kernel<<<qkv_blocks, blk, 0, stream>>>(
        prevb, wqt[l], bq + (size_t)l * HT, Qbf, nullptr, nullptr, scal + 0, nullptr,
        NROWS, HT, HIDD, 0);
    gemm_qkv_kernel<<<qkv_blocks, blk, 0, stream>>>(
        prevb, wkt[l], bk + (size_t)l * HT, nullptr, Ktb, nullptr, scal + 1, ksum,
        NROWS, HT, HIDD, 1);
    gemm_qkv_kernel<<<qkv_blocks, blk, 0, stream>>>(
        prevb, wvt[l], bv + (size_t)l * HT, nullptr, Vtb, Vf, nullptr, nullptr,
        NROWS, HT, HIDD, 2);

    kv_gemm_kernel<<<(2 * 8 * 32 * 16) / 8, blk, 0, stream>>>(Vtb, Ktb, kvtf, NROWS, 16);
    finalize_inv_kernel<<<1, 32, 0, stream>>>(scal);
    cvt_f32_bf16_kernel<<<2048, blk, 0, stream>>>(kvtf, kvtb, (size_t)2 * HIDD * HIDD);
    denom_kernel<<<(NROWS * 2) / 8, blk, 0, stream>>>(Qbf, ksum, scal, denom, NROWS);
    attn_final_kernel<<<(NROWS / 16) * (HIDD / 64) / 8, blk, 0, stream>>>(
        Qbf, kvtb, Vf, denom, scal, prevf, tmp, NROWS);

    if (l == 0) {
      layernorm_kernel<<<NROWS / 8, blk, 0, stream>>>(
          tmp, ln_g, ln_b, prevB, prevB_bf, 0, NROWS);
      prevf = prevB;
      prevb = prevB_bf;
    } else {
      layernorm_kernel<<<NROWS / 8, blk, 0, stream>>>(
          tmp, ln_g + HIDD, ln_b + HIDD, (float*)d_out, nullptr, 0, NROWS);
    }
  }
}
